// MultiHeadAttention_39711267619188
// MI455X (gfx1250) — compile-verified, matrix operand order checked by reading, not run on hardware
//
#include <hip/hip_runtime.h>

// ---------------------------------------------------------------------------
// MI455X (gfx1250) multi-head attention, bf16 WMMA pipeline.
//   stage 1: fp32 -> bf16 conversions (weights + activations)
//   stage 2: projection GEMMs, 32x64 tiles  (v_wmma_f32_16x16x32_bf16)
//   stage 3: flash-attention, 64-key blocks (v_wmma + exp2-domain softmax)
//   stage 4: output projection (fp32 out)
// ---------------------------------------------------------------------------

typedef unsigned short u16;
typedef __attribute__((ext_vector_type(8)))  u16    u16x8;
typedef __attribute__((ext_vector_type(16))) u16    u16x16;
typedef __attribute__((ext_vector_type(8)))  float  v8f;
typedef __attribute__((ext_vector_type(16))) __bf16 v16bf;

#define DEV static __device__ __forceinline__

// Problem constants (fixed by the reference).
constexpr int Bsz = 4;
constexpr int S   = 2048;
constexpr int D   = 768;
constexpr int H   = 12;
constexpr int DK  = 64;
constexpr int M   = Bsz * S;   // 8192 rows for all dense GEMMs

DEV u16 f2bf(float f) {             // fp32 -> bf16, round-to-nearest-even
  unsigned u = __builtin_bit_cast(unsigned, f);
  u += 0x7FFFu + ((u >> 16) & 1u);
  return (u16)(u >> 16);
}

// 16-bit A/B fragment load (wave32). `row` points at 32 contiguous bf16
// covering this fragment's K-chunk. Per the CDNA5 layout, lane (sel = l>>4)
// holds K offsets sel*8+{0..7} and 16+sel*8+{0..7} -> two b128 loads.
DEV u16x16 load_frag(const u16* row, int sel) {
  u16x8 lo = *(const u16x8*)(row + sel * 8);
  u16x8 hi = *(const u16x8*)(row + 16 + sel * 8);
  return __builtin_shufflevector(lo, hi, 0, 1, 2, 3, 4, 5, 6, 7,
                                 8, 9, 10, 11, 12, 13, 14, 15);
}

DEV v8f wmma_bf16(u16x16 a, u16x16 b, v8f c) {
  return __builtin_amdgcn_wmma_f32_16x16x32_bf16(
      false, __builtin_bit_cast(v16bf, a),
      false, __builtin_bit_cast(v16bf, b),
      (short)0, c, false, false);
}

DEV float rowmax16(float v) {
#pragma unroll
  for (int m = 1; m < 16; m <<= 1) v = fmaxf(v, __shfl_xor(v, m, 32));
  return v;
}
DEV float rowsum16(float v) {
#pragma unroll
  for (int m = 1; m < 16; m <<= 1) v += __shfl_xor(v, m, 32);
  return v;
}

// ---------------------------------------------------------------------------
// fp32 -> bf16 conversion
// ---------------------------------------------------------------------------
__global__ void f32_to_bf16_kernel(const float* __restrict__ in,
                                   u16* __restrict__ out, int n) {
  int i = blockIdx.x * blockDim.x + threadIdx.x;
  int stride = gridDim.x * blockDim.x;
  for (; i < n; i += stride) out[i] = f2bf(in[i]);
}

// ---------------------------------------------------------------------------
// GEMM: Y[m,n] = sum_k A[m,k] * W[n,k] + bias[n]
//   A : [M, 768] bf16 row-major          W : [768, 768] bf16 (torch [out,in])
// Each wave owns a 32x64 output tile: 2 A-fragments x 4 B-fragments -> 8 WMMA
// per K step on 12 b128 loads (1.5 loads/WMMA).
// mode 0: bf16 out, head-split [b,h,s,d]        (q and k projections)
// mode 1: bf16 out, transposed [b,h,d,s]        (v projection)
// mode 2: fp32 out, [M, 768]                    (final projection)
// ---------------------------------------------------------------------------
__global__ __launch_bounds__(128)
void gemm_bf16_kernel(const u16* __restrict__ A, const u16* __restrict__ W,
                      const float* __restrict__ bias, u16* __restrict__ outb,
                      float* __restrict__ outf, int mode) {
  const int wave = blockIdx.x * (blockDim.x >> 5) + (threadIdx.x >> 5);
  const int ntg  = D / 64;                       // 12 strips of 64 columns
  const int tm   = wave / ntg;                   // 32-row tile index
  const int tg   = wave % ntg;
  if (tm * 32 >= M) return;

  const int l = threadIdx.x & 31, sel = l >> 4, ln = l & 15;
  const u16* arow0 = A + (size_t)(tm * 32 + ln) * D;
  const u16* arow1 = A + (size_t)(tm * 32 + 16 + ln) * D;

  v8f acc[2][4] = {};
  for (int k0 = 0; k0 < D; k0 += 32) {
    u16x16 af0 = load_frag(arow0 + k0, sel);
    u16x16 af1 = load_frag(arow1 + k0, sel);
#pragma unroll
    for (int t = 0; t < 4; t++) {
      const u16* wrow = W + (size_t)(tg * 64 + t * 16 + ln) * D + k0;
      u16x16 bf = load_frag(wrow, sel);
      acc[0][t] = wmma_bf16(af0, bf, acc[0][t]);
      acc[1][t] = wmma_bf16(af1, bf, acc[1][t]);
    }
  }

#pragma unroll
  for (int r = 0; r < 2; r++) {
    const int Mi0 = tm * 32 + r * 16;
#pragma unroll
    for (int t = 0; t < 4; t++) {
      const int ng = tg * 64 + t * 16 + ln;      // lane's output column
      const float bv = bias[ng];
      if (mode == 2) {                           // fp32 [M, D]
#pragma unroll
        for (int i = 0; i < 8; i++) {
          int Mi = Mi0 + i + 8 * sel;            // C layout: row = i + 8*sel
          outf[(size_t)Mi * D + ng] = acc[r][t][i] + bv;
        }
      } else if (mode == 0) {                    // bf16 [b,h,s,d]
        int h = ng >> 6, d = ng & 63;
#pragma unroll
        for (int i = 0; i < 8; i++) {
          int Mi = Mi0 + i + 8 * sel;
          int b = Mi >> 11, s = Mi & (S - 1);
          outb[(((size_t)(b * H + h)) * S + s) * DK + d] =
              f2bf(acc[r][t][i] + bv);
        }
      } else {                                   // bf16 [b,h,d,s] (transposed)
        int h = ng >> 6, d = ng & 63;
        int Mi = Mi0 + 8 * sel;                  // 8 consecutive s per lane
        int b = Mi >> 11, s0 = Mi & (S - 1);
        u16x8 pack;
#pragma unroll
        for (int i = 0; i < 8; i++) pack[i] = f2bf(acc[r][t][i] + bv);
        *(u16x8*)(outb + (((size_t)(b * H + h)) * DK + d) * S + s0) = pack;
      }
    }
  }
}

// ---------------------------------------------------------------------------
// Flash attention. One wave per (b, h, 16-query tile); streams 64-key blocks
// (amortizes the softmax shuffle-reductions and accumulator rescale; WMMA
// co-executes with the VALU softmax stream).
//   Q,K : [b,h,s,d] bf16      Vt : [b,h,d,s] bf16     out: bf16 [B*S, 768]
// Softmax runs in the exp2 domain: scores are scaled by log2(e)/sqrt(dk), so
// every exponential is a single native v_exp_f32.
// ---------------------------------------------------------------------------
__global__ __launch_bounds__(128)
void attn_kernel(const u16* __restrict__ Q, const u16* __restrict__ K,
                 const u16* __restrict__ Vt, u16* __restrict__ Obf) {
  __shared__ __align__(16) u16 ldsP[4][16 * 64];   // 2 KB per wave

  const int wave = blockIdx.x * (blockDim.x >> 5) + (threadIdx.x >> 5);
  const int nqt  = S / 16;                         // 128 query tiles
  const int qt   = wave % nqt;
  const int bh   = wave / nqt;                     // b*H + h, [0,48)
  const int l = threadIdx.x & 31, sel = l >> 4, ln = l & 15;
  u16* myP = ldsP[threadIdx.x >> 5];

  const u16* Qb = Q  + (size_t)bh * S * DK;
  const u16* Kb = K  + (size_t)bh * S * DK;
  const u16* Vb = Vt + (size_t)bh * DK * S;

  // Q fragments for both 32-wide d chunks (reused all iterations).
  const u16* qrow = Qb + (size_t)(qt * 16 + ln) * DK;
  u16x16 qf0 = load_frag(qrow, sel);
  u16x16 qf1 = load_frag(qrow + 32, sel);

  // exp2-domain score scale: log2(e) / sqrt(dk)
  const float sc = 0.125f * 1.44269504088896340736f;

  v8f accO[4] = {};
  float mrow[8], lrow[8];
#pragma unroll
  for (int i = 0; i < 8; i++) { mrow[i] = -__builtin_inff(); lrow[i] = 0.f; }

  for (int j = 0; j < S / 64; j++) {
    // ---- scores for keys [j*64, j*64+64) : four 16-key C tiles ----
    v8f c[4];
#pragma unroll
    for (int kt = 0; kt < 4; kt++) {
      const u16* krow = Kb + (size_t)(j * 64 + kt * 16 + ln) * DK;
      v8f ct = {};
      ct = wmma_bf16(qf0, load_frag(krow, sel), ct);
      ct = wmma_bf16(qf1, load_frag(krow + 32, sel), ct);
      c[kt] = ct;
    }

    // ---- online softmax update (exp2 domain), write P tile to LDS ----
#pragma unroll
    for (int i = 0; i < 8; i++) {
      float s0 = c[0][i] * sc, s1 = c[1][i] * sc;
      float s2 = c[2][i] * sc, s3 = c[3][i] * sc;
      float rm = rowmax16(fmaxf(fmaxf(s0, s1), fmaxf(s2, s3)));
      float nm = fmaxf(mrow[i], rm);
      float alpha = exp2f(mrow[i] - nm);
      float p0 = exp2f(s0 - nm), p1 = exp2f(s1 - nm);
      float p2 = exp2f(s2 - nm), p3 = exp2f(s3 - nm);
      float rs = rowsum16((p0 + p1) + (p2 + p3));
      lrow[i] = lrow[i] * alpha + rs;
      mrow[i] = nm;
#pragma unroll
      for (int t = 0; t < 4; t++) accO[t][i] *= alpha;
      int row = i + 8 * sel;                     // C layout row
      myP[row * 64 + ln]      = f2bf(p0);
      myP[row * 64 + 16 + ln] = f2bf(p1);
      myP[row * 64 + 32 + ln] = f2bf(p2);
      myP[row * 64 + 48 + ln] = f2bf(p3);
    }
    __syncthreads();                             // uniform trip count: safe

    // ---- accO += P(16x64) x V(64x64) ----
    u16x16 pf0 = load_frag(myP + ln * 64, sel);  // keys j*64 +  0..31
    u16x16 pf1 = load_frag(myP + ln * 64 + 32, sel); // keys +32..63
#pragma unroll
    for (int t = 0; t < 4; t++) {
      const u16* vrow = Vb + (size_t)(t * 16 + ln) * S + j * 64;
      accO[t] = wmma_bf16(pf0, load_frag(vrow, sel), accO[t]);
      accO[t] = wmma_bf16(pf1, load_frag(vrow + 32, sel), accO[t]);
    }
    __syncthreads();
  }

  // ---- normalize and store to [B*S, 768] (col = h*64 + d) ----
  const int b = bh / H, h = bh % H;
#pragma unroll
  for (int i = 0; i < 8; i++) {
    float inv = 1.f / lrow[i];
    int s = qt * 16 + i + 8 * sel;
    size_t rowoff = ((size_t)b * S + s) * D + h * 64;
#pragma unroll
    for (int t = 0; t < 4; t++)
      Obf[rowoff + t * 16 + ln] = f2bf(accO[t][i] * inv);
  }
}

// ---------------------------------------------------------------------------
// Host orchestration (all launches on `stream`; graph-capture safe).
// ---------------------------------------------------------------------------
extern "C" void kernel_launch(void* const* d_in, const int* in_sizes, int n_in,
                              void* d_out, int out_size, void* d_ws,
                              size_t ws_size, hipStream_t stream) {
  const float* query = (const float*)d_in[0];
  const float* key   = (const float*)d_in[1];
  const float* value = (const float*)d_in[2];
  const float* Wq    = (const float*)d_in[3];
  const float* bq    = (const float*)d_in[4];
  const float* Wo    = (const float*)d_in[5];
  const float* bo    = (const float*)d_in[6];
  float* out = (float*)d_out;

  // Workspace sub-allocation (256B aligned).
  char* ws = (char*)d_ws;
  size_t off = 0;
  auto suballoc = [&](size_t bytes) -> void* {
    void* p = ws + off;
    off = (off + bytes + 255) & ~(size_t)255;
    return p;
  };
  const size_t actBytes = (size_t)M * D * sizeof(u16);   // 12.6 MB
  const size_t wBytes   = (size_t)D * D * sizeof(u16);   // 1.2 MB
  u16* xbf    = (u16*)suballoc(actBytes);   // staging for query/key/value
  u16* qbf    = (u16*)suballoc(actBytes);   // [b,h,s,d]
  u16* kbf    = (u16*)suballoc(actBytes);   // [b,h,s,d]
  u16* vtbf   = (u16*)suballoc(actBytes);   // [b,h,d,s]
  u16* attnbf = (u16*)suballoc(actBytes);   // [B*S, 768]
  u16* wqbf   = (u16*)suballoc(wBytes);
  u16* wobf   = (u16*)suballoc(wBytes);

  const int convT = 256;
  const int nAct = M * D, nW = D * D;
  const int gemmBlocks = (M / 32) * (D / 64) / 4;        // 768 blocks, 4 waves
  const int attnBlocks = (Bsz * H * (S / 16)) / 4;       // 1536 blocks, 4 waves

  // Weights to bf16 (once per call).
  f32_to_bf16_kernel<<<(nW + convT - 1) / convT, convT, 0, stream>>>(Wq, wqbf, nW);
  f32_to_bf16_kernel<<<(nW + convT - 1) / convT, convT, 0, stream>>>(Wo, wobf, nW);

  // q = query @ Wq^T + bq  -> [b,h,s,d]
  f32_to_bf16_kernel<<<(nAct + convT - 1) / convT, convT, 0, stream>>>(query, xbf, nAct);
  gemm_bf16_kernel<<<gemmBlocks, 128, 0, stream>>>(xbf, wqbf, bq, qbf, nullptr, 0);

  // k = key @ Wq^T + bq    -> [b,h,s,d]   (reference reuses Wq/bq)
  f32_to_bf16_kernel<<<(nAct + convT - 1) / convT, convT, 0, stream>>>(key, xbf, nAct);
  gemm_bf16_kernel<<<gemmBlocks, 128, 0, stream>>>(xbf, wqbf, bq, kbf, nullptr, 0);

  // v = value @ Wq^T + bq  -> [b,h,d,s] transposed
  f32_to_bf16_kernel<<<(nAct + convT - 1) / convT, convT, 0, stream>>>(value, xbf, nAct);
  gemm_bf16_kernel<<<gemmBlocks, 128, 0, stream>>>(xbf, wqbf, bq, vtbf, nullptr, 1);

  // attention -> [B*S, 768] bf16
  attn_kernel<<<attnBlocks, 128, 0, stream>>>(qbf, kbf, vtbf, attnbf);

  // out = attn @ Wo^T + bo -> fp32 d_out
  gemm_bf16_kernel<<<gemmBlocks, 128, 0, stream>>>(attnbf, wobf, bo, nullptr, out, 2);
}
